// RBF_58128087384571
// MI455X (gfx1250) — matrix operand order repaired and checked by reading, not compiled, not run on hardware
//
#include <hip/hip_runtime.h>
#include <cstdint>

#define B_ROWS 8192
#define L_ROWS 2048
#define DDIM   256

typedef __attribute__((ext_vector_type(16))) __bf16 v16bf;
typedef __attribute__((ext_vector_type(8)))  float  v8f;

union FragU { uint4 u[2]; v16bf v; };
union PackU { __bf16 h[8]; uint4 u; };

// ---------------------------------------------------------------------------
// A-fragment (16x32 MxK, bf16): lane<16 holds row M=lane, K = {k0..k0+7,
// k0+16..k0+23}; lane>=16 holds row M=lane-16, K = {k0+8..k0+15, k0+24..k0+31}.
// Per lane: two contiguous 16B chunks.
__device__ __forceinline__ v16bf load_fragA(const __bf16* row, int k0, int laneHi) {
    const int c0 = k0 + (laneHi ? 8 : 0);
    FragU f;
    f.u[0] = *reinterpret_cast<const uint4*>(row + c0);
    f.u[1] = *reinterpret_cast<const uint4*>(row + c0 + 16);
    return f.v;
}

// B-fragment (32x16 KxN, bf16): lane<16 holds column N=lane, K = {k0..k0+7,
// k0+16..k0+23}; lane>=16 holds N=lane-16, K = {k0+8..k0+15, k0+24..k0+31}.
// B[k][n] = landmarks[n][k], so each lane reads two 16B chunks of its landmark row.
__device__ __forceinline__ v16bf load_fragB(const __bf16* row, int k0, int laneHi) {
    const int c0 = k0 + (laneHi ? 8 : 0);
    FragU f;
    f.u[0] = *reinterpret_cast<const uint4*>(row + c0);
    f.u[1] = *reinterpret_cast<const uint4*>(row + c0 + 16);
    return f.v;
}

// ---------------------------------------------------------------------------
// fp32 -> (bf16 hi, bf16 lo) split + fp32 row squared-norm.
// One wave32 per row of D=256: 8 consecutive floats per lane.
__global__ __launch_bounds__(256) void rbf_cvt(const float* __restrict__ src,
                                               __bf16* __restrict__ hi,
                                               __bf16* __restrict__ lo,
                                               float* __restrict__ nrm) {
    const int lane = threadIdx.x & 31;
    const int row  = blockIdx.x * 8 + (threadIdx.x >> 5);
    const size_t base = (size_t)row * DDIM + (size_t)lane * 8;

    const float4 a = *reinterpret_cast<const float4*>(src + base);
    const float4 b = *reinterpret_cast<const float4*>(src + base + 4);
    float v[8] = {a.x, a.y, a.z, a.w, b.x, b.y, b.z, b.w};

    float s = 0.0f;
    PackU ph, pl;
#pragma unroll
    for (int i = 0; i < 8; ++i) {
        s += v[i] * v[i];
        __bf16 h = (__bf16)v[i];
        ph.h[i] = h;
        pl.h[i] = (__bf16)(v[i] - (float)h);
    }
    *reinterpret_cast<uint4*>(hi + base) = ph.u;
    *reinterpret_cast<uint4*>(lo + base) = pl.u;

#pragma unroll
    for (int off = 16; off > 0; off >>= 1) s += __shfl_xor(s, off);
    if (lane == 0) nrm[row] = s;
}

// ---------------------------------------------------------------------------
// Main GEMM + RBF epilogue. Block = 256 threads = 8 waves (2 x 4).
// WG tile: 128 (M/B-rows) x 256 (N/L-cols); each wave: 64x64 via 4x4 WMMA
// tiles. cross accumulated in f32 from bf16 hi/lo split (3 WMMAs per tile).
__global__ __launch_bounds__(256) void rbf_wmma(
    const __bf16* __restrict__ XH, const __bf16* __restrict__ XL,
    const __bf16* __restrict__ LH, const __bf16* __restrict__ LL,
    const float* __restrict__ X2, const float* __restrict__ L2N,
    float* __restrict__ out) {
    const float gamma = 1.0f / (float)DDIM;

    const int tid    = threadIdx.x;
    const int lane   = tid & 31;
    const int laneHi = (lane >> 4) & 1;
    const int lmod   = lane & 15;
    const int wave   = tid >> 5;
    const int waveM  = wave >> 2;  // 0..1
    const int waveN  = wave & 3;   // 0..3

    const int mBase = blockIdx.x * 128 + waveM * 64;
    const int nBase = blockIdx.y * 256 + waveN * 64;

    size_t aOff[4], bOff[4];
#pragma unroll
    for (int t = 0; t < 4; ++t) {
        aOff[t] = (size_t)(mBase + t * 16 + lmod) * DDIM;
        bOff[t] = (size_t)(nBase + t * 16 + lmod) * DDIM;
    }

    v8f acc[4][4];
    const v8f vzero = {0.f, 0.f, 0.f, 0.f, 0.f, 0.f, 0.f, 0.f};
#pragma unroll
    for (int mt = 0; mt < 4; ++mt)
#pragma unroll
        for (int nt = 0; nt < 4; ++nt) acc[mt][nt] = vzero;

#pragma unroll 1
    for (int k0 = 0; k0 < DDIM; k0 += 32) {
        v16bf ah[4], al[4], bh[4], bl[4];
#pragma unroll
        for (int t = 0; t < 4; ++t) {
            ah[t] = load_fragA(XH + aOff[t], k0, laneHi);
            al[t] = load_fragA(XL + aOff[t], k0, laneHi);
            bh[t] = load_fragB(LH + bOff[t], k0, laneHi);
            bl[t] = load_fragB(LL + bOff[t], k0, laneHi);
        }
#pragma unroll
        for (int mt = 0; mt < 4; ++mt) {
#pragma unroll
            for (int nt = 0; nt < 4; ++nt) {
                acc[mt][nt] = __builtin_amdgcn_wmma_f32_16x16x32_bf16(
                    false, ah[mt], false, bh[nt], (short)0, acc[mt][nt], false, false);
                acc[mt][nt] = __builtin_amdgcn_wmma_f32_16x16x32_bf16(
                    false, al[mt], false, bh[nt], (short)0, acc[mt][nt], false, false);
                acc[mt][nt] = __builtin_amdgcn_wmma_f32_16x16x32_bf16(
                    false, ah[mt], false, bl[nt], (short)0, acc[mt][nt], false, false);
            }
        }
    }

    // Epilogue: out[r, c] = exp(-gamma * max(x2[r] + l2[c] - 2*cross, 0)).
    // C/D 16x16 f32 layout: VGPR j -> row M = j (+8 for lanes>=16), lane -> N.
    float l2v[4];
    int   ccol[4];
#pragma unroll
    for (int nt = 0; nt < 4; ++nt) {
        ccol[nt] = nBase + nt * 16 + lmod;
        l2v[nt]  = L2N[ccol[nt]];
    }

#pragma unroll
    for (int mt = 0; mt < 4; ++mt) {
#pragma unroll
        for (int j = 0; j < 8; ++j) {
            const int r = mBase + mt * 16 + laneHi * 8 + j;
            const float xn = X2[r];
            float* orow = out + (size_t)r * L_ROWS;
#pragma unroll
            for (int nt = 0; nt < 4; ++nt) {
                float d2 = xn + l2v[nt] - 2.0f * acc[mt][nt][j];
                d2 = fmaxf(d2, 0.0f);
                orow[ccol[nt]] = __expf(-gamma * d2);
            }
        }
    }
}

// ---------------------------------------------------------------------------
extern "C" void kernel_launch(void* const* d_in, const int* in_sizes, int n_in,
                              void* d_out, int out_size, void* d_ws, size_t ws_size,
                              hipStream_t stream) {
    (void)in_sizes; (void)n_in; (void)out_size; (void)ws_size;
    const float* x  = (const float*)d_in[0];   // [B, D] f32
    const float* lm = (const float*)d_in[1];   // [L, D] f32
    float* out = (float*)d_out;                // [B, L] f32

    // Workspace layout (all 16B-aligned): bf16 hi/lo planes + f32 norms.
    uint16_t* w = (uint16_t*)d_ws;
    uint16_t* xh = w;
    uint16_t* xl = xh + (size_t)B_ROWS * DDIM;
    uint16_t* lh = xl + (size_t)B_ROWS * DDIM;
    uint16_t* ll = lh + (size_t)L_ROWS * DDIM;
    float*    x2 = (float*)(ll + (size_t)L_ROWS * DDIM);
    float*    l2 = x2 + B_ROWS;

    rbf_cvt<<<B_ROWS / 8, 256, 0, stream>>>(x,  (__bf16*)xh, (__bf16*)xl, x2);
    rbf_cvt<<<L_ROWS / 8, 256, 0, stream>>>(lm, (__bf16*)lh, (__bf16*)ll, l2);

    dim3 grid(B_ROWS / 128, L_ROWS / 256);  // 64 x 8 workgroups
    rbf_wmma<<<grid, 256, 0, stream>>>((const __bf16*)xh, (const __bf16*)xl,
                                       (const __bf16*)lh, (const __bf16*)ll,
                                       x2, l2, out);
}
